// AttentionBlock3D_14989435863547
// MI455X (gfx1250) — matrix operand order repaired and checked by reading, not compiled, not run on hardware
//
#include <hip/hip_runtime.h>
#include <hip/hip_bf16.h>

// ---------------------------------------------------------------------------
// AttentionBlock3D for MI455X (gfx1250): GroupNorm -> QKV GEMM -> flash
// attention (NH=8, hd=32, S=4096) -> proj GEMM + residual.
// All GEMMs via v_wmma_f32_16x16x32_bf16 (wave32). 2x2 register blocking per
// wave (32x32 output) for 2x fragment reuse; attention does 32 queries/wave.
// ---------------------------------------------------------------------------

typedef __bf16 bf16;
typedef __attribute__((ext_vector_type(16))) __bf16 v16bf;
typedef __attribute__((ext_vector_type(8)))  float  v8f;

constexpr int Bn  = 2;
constexpr int Cn  = 256;
constexpr int NHn = 8;
constexpr int Gn  = 32;
constexpr int Sn  = 4096;       // 16*16*16
constexpr int HDn = 32;         // head dim
constexpr int CPG = Cn / Gn;    // 8 channels per group

__device__ __forceinline__ v8f wmma_bf16(v16bf a, v16bf b, v8f c) {
  return __builtin_amdgcn_wmma_f32_16x16x32_bf16(
      /*neg_a=*/false, a, /*neg_b=*/false, b,
      /*c_mod=*/(short)0, c, /*reuse_a=*/false, /*reuse_b=*/false);
}

// A fragment: 16x32 bf16, row-major, row stride = ld. base -> element (m=0,k=0).
// Lane L holds row m=L%16; VGPR v holds K=(L/16)*16 + 2v, +1.
__device__ __forceinline__ v16bf load_frag_a(const bf16* base, int ld, int lane) {
  int m = lane & 15, half = lane >> 4;
  const bf16* rp = base + (size_t)m * ld;
  v16bf a;
#pragma unroll
  for (int v = 0; v < 8; ++v) {
    int kk = (half << 4) + (v << 1);
    a[2 * v]     = rp[kk];
    a[2 * v + 1] = rp[kk + 1];
  }
  return a;
}

// B fragment: 32x16 bf16, stored K-contiguous per column: element (k,n) at
// base + n*col_stride + k. Lane L holds col n=L%16; VGPR v holds
// K=(L/16)*16 + 2v, +1.
__device__ __forceinline__ v16bf load_frag_b(const bf16* base, int col_stride, int lane) {
  int n = lane & 15, half = lane >> 4;
  const bf16* cp = base + (size_t)n * col_stride + (half << 4);
  v16bf b;
#pragma unroll
  for (int v = 0; v < 8; ++v) {
    b[2 * v]     = cp[2 * v];
    b[2 * v + 1] = cp[2 * v + 1];
  }
  return b;
}

// ---------------------------------------------------------------------------
// Kernel 0: convert weights f32 -> bf16
// ---------------------------------------------------------------------------
__global__ void ab3d_convert_w(const float* __restrict__ qw, const float* __restrict__ pw,
                               bf16* __restrict__ wq, bf16* __restrict__ wp) {
  int i = blockIdx.x * 256 + threadIdx.x;
  if (i < 3 * Cn * Cn) wq[i] = (bf16)qw[i];
  if (i < Cn * Cn)     wp[i] = (bf16)pw[i];
}

// ---------------------------------------------------------------------------
// Kernel 1: GroupNorm over (C/G * S) elems per (b,g); writes h transposed
// as (B, S, C) bf16 so the QKV GEMM B-fragments are K-contiguous.
// ---------------------------------------------------------------------------
__global__ void ab3d_groupnorm(const float* __restrict__ x,
                               const float* __restrict__ gamma,
                               const float* __restrict__ beta,
                               bf16* __restrict__ h_t) {
  const int g = blockIdx.x, b = blockIdx.y;
  const int N = CPG * Sn;                       // 32768
  const float* xp = x + ((size_t)b * Cn + (size_t)g * CPG) * Sn;
  __shared__ float red[256];

  float sum = 0.f, sq = 0.f;
  for (int i = threadIdx.x; i < N; i += 256) {
    float v = xp[i];
    sum += v; sq += v * v;
  }
  red[threadIdx.x] = sum; __syncthreads();
  for (int o = 128; o > 0; o >>= 1) {
    if (threadIdx.x < (unsigned)o) red[threadIdx.x] += red[threadIdx.x + o];
    __syncthreads();
  }
  float tot = red[0]; __syncthreads();
  red[threadIdx.x] = sq; __syncthreads();
  for (int o = 128; o > 0; o >>= 1) {
    if (threadIdx.x < (unsigned)o) red[threadIdx.x] += red[threadIdx.x + o];
    __syncthreads();
  }
  float mean = tot / (float)N;
  float var  = red[0] / (float)N - mean * mean;
  float rstd = rsqrtf(var + 1e-6f);

  for (int i = threadIdx.x; i < N; i += 256) {
    int c = g * CPG + i / Sn;
    int s = i % Sn;
    float v = (xp[i] - mean) * rstd * gamma[c] + beta[c];
    h_t[((size_t)b * Sn + s) * Cn + c] = (bf16)v;
  }
}

// ---------------------------------------------------------------------------
// Kernel 2: QKV GEMM, 2x2 blocked. qkv[o,s] = sum_c W[o,c]*h[c,s] + bias[o].
// One wave per 32x32 tile; K=256 -> 8 iters x 4 WMMAs, 2 A + 2 B frags/iter.
// Scatter into head layouts:
//   q,k : (B, NH, S, hd)  row-major (queries/keys K-contiguous as A/B frags)
//   v_t : (B, NH, hd, S)  transposed (PV B-fragments K-contiguous)
// ---------------------------------------------------------------------------
__global__ void ab3d_qkv_gemm(const bf16* __restrict__ w, const float* __restrict__ bias,
                              const bf16* __restrict__ h_t,
                              bf16* __restrict__ q, bf16* __restrict__ k,
                              bf16* __restrict__ v_t) {
  const int lane = threadIdx.x;
  const int s0 = blockIdx.x * 32;
  const int m0 = blockIdx.y * 32;          // 0..767
  const int b  = blockIdx.z;
  const bf16* hb = h_t + (size_t)b * Sn * Cn;

  v8f acc00 = {}, acc01 = {}, acc10 = {}, acc11 = {};
#pragma unroll
  for (int kk = 0; kk < Cn; kk += 32) {
    if (kk + 32 < Cn) {
      __builtin_prefetch(hb + (size_t)s0 * Cn + kk + 32, 0, 1);
      __builtin_prefetch(hb + (size_t)(s0 + 16) * Cn + kk + 32, 0, 1);
    }
    v16bf a0 = load_frag_a(w + (size_t)m0 * Cn + kk, Cn, lane);
    v16bf a1 = load_frag_a(w + (size_t)(m0 + 16) * Cn + kk, Cn, lane);
    v16bf b0 = load_frag_b(hb + (size_t)s0 * Cn + kk, Cn, lane);
    v16bf b1 = load_frag_b(hb + (size_t)(s0 + 16) * Cn + kk, Cn, lane);
    acc00 = wmma_bf16(a0, b0, acc00);
    acc01 = wmma_bf16(a0, b1, acc01);
    acc10 = wmma_bf16(a1, b0, acc10);
    acc11 = wmma_bf16(a1, b1, acc11);
  }

  const int nl = lane & 15, half = lane >> 4;
#pragma unroll
  for (int mi = 0; mi < 2; ++mi) {
#pragma unroll
    for (int ni = 0; ni < 2; ++ni) {
      const v8f& acc = mi == 0 ? (ni == 0 ? acc00 : acc01)
                               : (ni == 0 ? acc10 : acc11);
      int s = s0 + ni * 16 + nl;
#pragma unroll
      for (int r = 0; r < 8; ++r) {
        int m = m0 + mi * 16 + r + (half << 3);
        float val = acc[r] + bias[m];
        int oc = m & 255;
        int which = m >> 8;                // 0=q 1=k 2=v
        int nh = oc >> 5, d = oc & 31;
        if (which == 0)
          q[(((size_t)b * NHn + nh) * Sn + s) * HDn + d] = (bf16)val;
        else if (which == 1)
          k[(((size_t)b * NHn + nh) * Sn + s) * HDn + d] = (bf16)val;
        else
          v_t[(((size_t)b * NHn + nh) * HDn + d) * Sn + s] = (bf16)val;
      }
    }
  }
}

// ---------------------------------------------------------------------------
// Kernel 3: flash attention. One wave per (b, head, 32-query tile).
// Per 32-key chunk: 4 score WMMAs (K=hd=32 in one shot, k-frags reused by
// both q tiles), online softmax in registers (shfl_xor row reductions within
// each 16-lane half), P re-striped C-layout -> A-layout through 2KB LDS,
// 4 PV WMMAs (v-frags reused by both P tiles).
// ---------------------------------------------------------------------------
__global__ void ab3d_attention(const bf16* __restrict__ q, const bf16* __restrict__ k,
                               const bf16* __restrict__ v_t, bf16* __restrict__ o_t) {
  const int lane = threadIdx.x;
  const int s0 = blockIdx.x * 32;
  const int nh = blockIdx.y, b = blockIdx.z;
  const bf16* qh = q  + (((size_t)b * NHn + nh) * Sn) * HDn;
  const bf16* kh = k  + (((size_t)b * NHn + nh) * Sn) * HDn;
  const bf16* vh = v_t + (((size_t)b * NHn + nh) * HDn) * Sn;

  __shared__ bf16 plds[32 * 32];

  v16bf qa[2];
  qa[0] = load_frag_a(qh + (size_t)s0 * HDn, HDn, lane);
  qa[1] = load_frag_a(qh + (size_t)(s0 + 16) * HDn, HDn, lane);
  v8f o00 = {}, o01 = {}, o10 = {}, o11 = {};
  float mrow[2][8], lrow[2][8];
#pragma unroll
  for (int mi = 0; mi < 2; ++mi)
#pragma unroll
    for (int r = 0; r < 8; ++r) { mrow[mi][r] = -3.0e38f; lrow[mi][r] = 0.f; }

  const float scale = 0.17677669529663687f;   // hd^-0.5
  const int half = lane >> 4, nl = lane & 15;

  for (int kb = 0; kb < Sn; kb += 32) {
    if (kb + 32 < Sn) {
      __builtin_prefetch(kh + (size_t)(kb + 32) * HDn, 0, 1);
      __builtin_prefetch(vh + (size_t)kb + 32, 0, 1);
    }
    v16bf kb0 = load_frag_b(kh + (size_t)kb * HDn, HDn, lane);
    v16bf kb1 = load_frag_b(kh + (size_t)(kb + 16) * HDn, HDn, lane);
    v8f sA[2], sB[2];
#pragma unroll
    for (int mi = 0; mi < 2; ++mi) {
      v8f z = {};
      sA[mi] = wmma_bf16(qa[mi], kb0, z);
      sB[mi] = wmma_bf16(qa[mi], kb1, z);
    }

#pragma unroll
    for (int mi = 0; mi < 2; ++mi) {
      float cm[8];
#pragma unroll
      for (int r = 0; r < 8; ++r) {
        sA[mi][r] *= scale; sB[mi][r] *= scale;
        cm[r] = fmaxf(sA[mi][r], sB[mi][r]);
      }
#pragma unroll
      for (int msk = 1; msk < 16; msk <<= 1)
#pragma unroll
        for (int r = 0; r < 8; ++r) cm[r] = fmaxf(cm[r], __shfl_xor(cm[r], msk, 32));

      float alpha[8], rs[8];
#pragma unroll
      for (int r = 0; r < 8; ++r) {
        float mn = fmaxf(mrow[mi][r], cm[r]);
        alpha[r] = __expf(mrow[mi][r] - mn);
        mrow[mi][r] = mn;
        float p0 = __expf(sA[mi][r] - mn);
        float p1 = __expf(sB[mi][r] - mn);
        sA[mi][r] = p0; sB[mi][r] = p1;
        rs[r] = p0 + p1;
      }
#pragma unroll
      for (int msk = 1; msk < 16; msk <<= 1)
#pragma unroll
        for (int r = 0; r < 8; ++r) rs[r] += __shfl_xor(rs[r], msk, 32);
#pragma unroll
      for (int r = 0; r < 8; ++r) {
        lrow[mi][r] = lrow[mi][r] * alpha[r] + rs[r];
        if (mi == 0) { o00[r] *= alpha[r]; o01[r] *= alpha[r]; }
        else         { o10[r] *= alpha[r]; o11[r] *= alpha[r]; }
      }

      // C-layout (m = r+8*half, n = nl) -> LDS row-major (32 rows x 32 keys)
#pragma unroll
      for (int r = 0; r < 8; ++r) {
        int m = mi * 16 + r + (half << 3);
        plds[m * 32 + nl]      = (bf16)sA[mi][r];
        plds[m * 32 + 16 + nl] = (bf16)sB[mi][r];
      }
    }
    __syncthreads();
    v16bf pa0 = load_frag_a(plds, 32, lane);
    v16bf pa1 = load_frag_a(plds + 16 * 32, 32, lane);
    v16bf vb0 = load_frag_b(vh + kb, Sn, lane);                   // dims 0..15
    v16bf vb1 = load_frag_b(vh + (size_t)16 * Sn + kb, Sn, lane); // dims 16..31
    o00 = wmma_bf16(pa0, vb0, o00);
    o01 = wmma_bf16(pa0, vb1, o01);
    o10 = wmma_bf16(pa1, vb0, o10);
    o11 = wmma_bf16(pa1, vb1, o11);
    __syncthreads();
  }

  // epilogue: divide by row sums, write (B, S, C) bf16 (K-contiguous for proj)
#pragma unroll
  for (int mi = 0; mi < 2; ++mi) {
#pragma unroll
    for (int r = 0; r < 8; ++r) {
      int m = r + (half << 3);
      int s = s0 + mi * 16 + m;
      float inv = 1.0f / lrow[mi][r];
      size_t base = ((size_t)b * Sn + s) * Cn + nh * HDn;
      float v0 = (mi == 0 ? o00[r] : o10[r]) * inv;
      float v1 = (mi == 0 ? o01[r] : o11[r]) * inv;
      o_t[base + nl]      = (bf16)v0;
      o_t[base + 16 + nl] = (bf16)v1;
    }
  }
}

// ---------------------------------------------------------------------------
// Kernel 4: proj GEMM + bias + residual, 2x2 blocked.
// out[b,o,s] = x + b[o] + W@o_head.
// ---------------------------------------------------------------------------
__global__ void ab3d_proj_gemm(const bf16* __restrict__ w, const float* __restrict__ bias,
                               const bf16* __restrict__ o_t, const float* __restrict__ x,
                               float* __restrict__ out) {
  const int lane = threadIdx.x;
  const int s0 = blockIdx.x * 32;
  const int m0 = blockIdx.y * 32;
  const int b  = blockIdx.z;
  const bf16* ob = o_t + (size_t)b * Sn * Cn;

  v8f acc00 = {}, acc01 = {}, acc10 = {}, acc11 = {};
#pragma unroll
  for (int kk = 0; kk < Cn; kk += 32) {
    if (kk + 32 < Cn) {
      __builtin_prefetch(ob + (size_t)s0 * Cn + kk + 32, 0, 1);
      __builtin_prefetch(ob + (size_t)(s0 + 16) * Cn + kk + 32, 0, 1);
    }
    v16bf a0 = load_frag_a(w + (size_t)m0 * Cn + kk, Cn, lane);
    v16bf a1 = load_frag_a(w + (size_t)(m0 + 16) * Cn + kk, Cn, lane);
    v16bf b0 = load_frag_b(ob + (size_t)s0 * Cn + kk, Cn, lane);
    v16bf b1 = load_frag_b(ob + (size_t)(s0 + 16) * Cn + kk, Cn, lane);
    acc00 = wmma_bf16(a0, b0, acc00);
    acc01 = wmma_bf16(a0, b1, acc01);
    acc10 = wmma_bf16(a1, b0, acc10);
    acc11 = wmma_bf16(a1, b1, acc11);
  }

  const int nl = lane & 15, half = lane >> 4;
#pragma unroll
  for (int mi = 0; mi < 2; ++mi) {
#pragma unroll
    for (int ni = 0; ni < 2; ++ni) {
      const v8f& acc = mi == 0 ? (ni == 0 ? acc00 : acc01)
                               : (ni == 0 ? acc10 : acc11);
      int s = s0 + ni * 16 + nl;
#pragma unroll
      for (int r = 0; r < 8; ++r) {
        int m = m0 + mi * 16 + r + (half << 3);
        size_t idx = ((size_t)b * Cn + m) * Sn + s;
        out[idx] = x[idx] + bias[m] + acc[r];
      }
    }
  }
}

// ---------------------------------------------------------------------------
extern "C" void kernel_launch(void* const* d_in, const int* in_sizes, int n_in,
                              void* d_out, int out_size, void* d_ws, size_t ws_size,
                              hipStream_t stream) {
  (void)in_sizes; (void)n_in; (void)out_size; (void)ws_size;
  const float* x      = (const float*)d_in[0];
  const float* gamma  = (const float*)d_in[1];
  const float* beta   = (const float*)d_in[2];
  const float* qkv_w  = (const float*)d_in[3];
  const float* qkv_b  = (const float*)d_in[4];
  const float* proj_w = (const float*)d_in[5];
  const float* proj_b = (const float*)d_in[6];
  float* out = (float*)d_out;

  char* ws = (char*)d_ws;
  size_t off = 0;
  auto alloc = [&](size_t bytes) -> void* {
    void* p = ws + off;
    off += (bytes + 255) & ~(size_t)255;
    return p;
  };
  bf16* h_t = (bf16*)alloc((size_t)Bn * Sn * Cn * sizeof(bf16));
  bf16* wq  = (bf16*)alloc((size_t)3 * Cn * Cn * sizeof(bf16));
  bf16* wp  = (bf16*)alloc((size_t)Cn * Cn * sizeof(bf16));
  bf16* qb  = (bf16*)alloc((size_t)Bn * NHn * Sn * HDn * sizeof(bf16));
  bf16* kb  = (bf16*)alloc((size_t)Bn * NHn * Sn * HDn * sizeof(bf16));
  bf16* vb  = (bf16*)alloc((size_t)Bn * NHn * HDn * Sn * sizeof(bf16));
  bf16* o_t = (bf16*)alloc((size_t)Bn * Sn * Cn * sizeof(bf16));

  // 0) weight conversion
  ab3d_convert_w<<<dim3((3 * Cn * Cn + 255) / 256), dim3(256), 0, stream>>>(
      qkv_w, proj_w, wq, wp);
  // 1) groupnorm -> h_t (B,S,C) bf16
  ab3d_groupnorm<<<dim3(Gn, Bn), dim3(256), 0, stream>>>(x, gamma, beta, h_t);
  // 2) qkv gemm -> q,k,(v transposed)
  ab3d_qkv_gemm<<<dim3(Sn / 32, (3 * Cn) / 32, Bn), dim3(32), 0, stream>>>(
      wq, qkv_b, h_t, qb, kb, vb);
  // 3) flash attention -> o_t (B,S,C) bf16
  ab3d_attention<<<dim3(Sn / 32, NHn, Bn), dim3(32), 0, stream>>>(qb, kb, vb, o_t);
  // 4) proj + residual -> out (B,C,S) f32
  ab3d_proj_gemm<<<dim3(Sn / 32, Cn / 32, Bn), dim3(32), 0, stream>>>(
      wp, proj_b, o_t, x, out);
}
